// MPT_53850299957772
// MI455X (gfx1250) — matrix operand order repaired and checked by reading, not compiled, not run on hardware
//
#include <hip/hip_runtime.h>

// ---------------------------------------------------------------------------
// Mamba2 forward for MI455X (gfx1250, wave32, WMMA).
// GEMMs: bf16 via v_wmma_f32_16x16x32_bf16, f32 accumulate.
// B tiles staged into LDS with global_load_async_to_lds_b128 (ASYNCcnt).
// ---------------------------------------------------------------------------

typedef unsigned short u16;
typedef __attribute__((ext_vector_type(16))) __bf16       v16bf;
typedef __attribute__((ext_vector_type(8)))  float        v8f;
typedef __attribute__((ext_vector_type(4)))  unsigned int uiv4;

union FragBF { uiv4 q[2]; v16bf v; };

#define D_MODEL    512
#define D_STATE    128
#define D_INNER    1024
#define HEADDIM    64
#define NHEADS     16
#define CONV_DIM   1280         // D_INNER + 2*D_STATE
#define D_PROJ     2320         // 2*D_INNER + 2*D_STATE + NHEADS
#define D_PROJ_PAD 2368         // padded to multiple of 64 (37*64)
#define N_LAYER    28
#define VOCAB      256
#define BATCH      16
#define SEQ        768
#define TOKENS     (BATCH * SEQ)   // 12288

__device__ __forceinline__ u16 f32_to_bf16(float f) {
  unsigned u = __float_as_uint(f);
  u += 0x7FFFu + ((u >> 16) & 1u);   // round-to-nearest-even
  return (u16)(u >> 16);
}

__device__ __forceinline__ float silu_f(float x) {
  return x / (1.f + __expf(-x));
}

// ---------------------------------------------------------------------------
// fp32 -> bf16 conversion kernels (weights; re-done every launch)
// ---------------------------------------------------------------------------
__global__ void cvt_bf16_kernel(const float* __restrict__ in,
                                u16* __restrict__ out, long long n) {
  long long i = (long long)blockIdx.x * blockDim.x + threadIdx.x;
  long long stride = (long long)gridDim.x * blockDim.x;
  for (; i < n; i += stride) out[i] = f32_to_bf16(in[i]);
}

// in_proj: [L, 2320, 512] f32 -> [L, 2368, 512] bf16, rows >= 2320 zeroed
__global__ void cvt_inproj_pad_kernel(const float* __restrict__ in,
                                      u16* __restrict__ out, long long n) {
  long long i = (long long)blockIdx.x * blockDim.x + threadIdx.x;
  long long stride = (long long)gridDim.x * blockDim.x;
  for (; i < n; i += stride) {
    int c = (int)(i & 511);
    long long rr = i >> 9;
    int r = (int)(rr % D_PROJ_PAD);
    int l = (int)(rr / D_PROJ_PAD);
    out[i] = (r < D_PROJ)
               ? f32_to_bf16(in[((size_t)l * D_PROJ + r) * D_MODEL + c])
               : (u16)0;
  }
}

// ---------------------------------------------------------------------------
// Embedding gather + zero residual
// ---------------------------------------------------------------------------
__global__ void embed_kernel(const int* __restrict__ tok,
                             const float* __restrict__ emb,
                             float* __restrict__ hidden,
                             float* __restrict__ res, int total) {
  int i = blockIdx.x * blockDim.x + threadIdx.x;
  int stride = gridDim.x * blockDim.x;
  for (; i < total; i += stride) {
    int t = i >> 9;           // / 512
    int c = i & 511;
    hidden[i] = emb[(size_t)tok[t] * D_MODEL + c];
    res[i] = 0.f;
  }
}

// ---------------------------------------------------------------------------
// res += h ; out_bf16 = bf16(rms(res) * w)      (one block per token)
// ---------------------------------------------------------------------------
__global__ __launch_bounds__(256) void addnorm_kernel(
    const float* __restrict__ h, float* __restrict__ res,
    const float* __restrict__ w, u16* __restrict__ out, int C) {
  const int t = blockIdx.x, tid = threadIdx.x;
  const float* hr = h + (size_t)t * C;
  float* rr = res + (size_t)t * C;
  float vals[4];
  const int per = C >> 8;
  float ss = 0.f;
  for (int i = 0; i < per; ++i) {
    int c = tid + (i << 8);
    float v = hr[c] + rr[c];
    rr[c] = v;
    vals[i] = v;
    ss = fmaf(v, v, ss);
  }
#pragma unroll
  for (int off = 16; off >= 1; off >>= 1) ss += __shfl_xor(ss, off, 32);
  __shared__ float sw[8];
  if ((tid & 31) == 0) sw[tid >> 5] = ss;
  __syncthreads();
  float tot = 0.f;
#pragma unroll
  for (int j = 0; j < 8; ++j) tot += sw[j];
  float rinv = rsqrtf(tot / (float)C + 1e-5f);
  for (int i = 0; i < per; ++i) {
    int c = tid + (i << 8);
    out[(size_t)t * C + c] = f32_to_bf16(vals[i] * rinv * w[c]);
  }
}

// ---------------------------------------------------------------------------
// Gated RMSNorm: v = y * silu(z); out = bf16(rms(v) * w)   (C = 1024)
// ---------------------------------------------------------------------------
__global__ __launch_bounds__(256) void gatenorm_kernel(
    const float* __restrict__ y, const float* __restrict__ zx,
    const float* __restrict__ w, u16* __restrict__ out) {
  const int t = blockIdx.x, tid = threadIdx.x;
  float vals[4];
  float ss = 0.f;
#pragma unroll
  for (int i = 0; i < 4; ++i) {
    int c = tid + (i << 8);
    float z = zx[(size_t)t * D_PROJ_PAD + c];
    float v = y[(size_t)t * D_INNER + c] * silu_f(z);
    vals[i] = v;
    ss = fmaf(v, v, ss);
  }
#pragma unroll
  for (int off = 16; off >= 1; off >>= 1) ss += __shfl_xor(ss, off, 32);
  __shared__ float sw[8];
  if ((tid & 31) == 0) sw[tid >> 5] = ss;
  __syncthreads();
  float tot = 0.f;
#pragma unroll
  for (int j = 0; j < 8; ++j) tot += sw[j];
  float rinv = rsqrtf(tot / 1024.f + 1e-5f);
#pragma unroll
  for (int i = 0; i < 4; ++i) {
    int c = tid + (i << 8);
    out[(size_t)t * D_INNER + c] = f32_to_bf16(vals[i] * rinv * w[c]);
  }
}

// ---------------------------------------------------------------------------
// bf16 GEMM: C[M,N] = A[M,K] * W[N,K]^T   (A,W bf16 row-major, C f32)
// Requires: M % 128 == 0, N % 64 == 0, K % 32 == 0  (padded by caller).
// Block = 256 threads (8 waves). Block tile 128(M) x 64(N), K step 32.
// B tile staged to LDS via async DMA (global_load_async_to_lds_b128).
// No divergent guards: EXEC is all-ones at every WMMA.
// ---------------------------------------------------------------------------
__global__ __launch_bounds__(256) void gemm_bf16_wmma(
    const u16* __restrict__ A, const u16* __restrict__ Bw,
    float* __restrict__ C, int N, int K) {
  __shared__ __align__(16) u16 sB[64 * 40];   // 64 rows x 32 K, 80B padded rows

  const int tid  = threadIdx.x;
  const int wave = tid >> 5;
  const int lane = tid & 31;
  const int m0w  = blockIdx.y * 128 + wave * 16;
  const int n0b  = blockIdx.x * 64;

  v8f acc[4];
#pragma unroll
  for (int j = 0; j < 4; ++j) acc[j] = (v8f)0.0f;

  const int srow = tid >> 2;          // 0..63 : staged B row (N within tile)
  const int sseg = tid & 3;           // 16B segment within 64B row
  const u16* brow_p = Bw + (size_t)(n0b + srow) * K + sseg * 8;
  const unsigned lds_dst =
      (unsigned)(size_t)(void*)&sB[srow * 40 + sseg * 8];

  const int aln = lane & 15;
  const int akb = (lane < 16) ? 0 : 8;    // A k-base per lane half
  const int bkb = (lane < 16) ? 0 : 8;    // B k-base per lane half
  const u16* arow_p = A + (size_t)(m0w + aln) * K;

  for (int kt = 0; kt < K; kt += 32) {
    __syncthreads();
    // Async DMA: 16B per lane, global -> LDS, tracked by ASYNCcnt.
    asm volatile("global_load_async_to_lds_b128 %0, %1, off"
                 :
                 : "v"(lds_dst), "v"(brow_p + kt)
                 : "memory");

    FragBF fa;
    const u16* ap = arow_p + kt + akb;
    fa.q[0] = *(const uiv4*)ap;          // k = kb..kb+7
    fa.q[1] = *(const uiv4*)(ap + 16);   // k = kb+16..kb+23
    __builtin_prefetch(ap + 32, 0, 0);   // speculative, fault-free

    asm volatile("s_wait_asynccnt 0x0" ::: "memory");
    __syncthreads();

#pragma unroll
    for (int j = 0; j < 4; ++j) {
      FragBF fb;
      const u16* bp = &sB[(j * 16 + aln) * 40 + bkb];
      fb.q[0] = *(const uiv4*)bp;        // k = kb..kb+7
      fb.q[1] = *(const uiv4*)(bp + 16); // k = kb+16..kb+23
      acc[j] = __builtin_amdgcn_wmma_f32_16x16x32_bf16(
          false, fa.v, false, fb.v, (short)0, acc[j], false, false);
    }
  }

  const int mrb = m0w + ((lane < 16) ? 0 : 8);
#pragma unroll
  for (int j = 0; j < 4; ++j) {
    const int ncol = n0b + j * 16 + aln;
#pragma unroll
    for (int r = 0; r < 8; ++r)
      C[(size_t)(mrb + r) * N + ncol] = acc[j][r];
  }
}

// ---------------------------------------------------------------------------
// Depthwise causal conv (K=4) + SiLU over xBC, and softplus(dt) in one pass.
// zx layout: [T, 2368] = [z(1024) | xBC(1280) | dt(16) | pad(48)]
// ---------------------------------------------------------------------------
__global__ void conv_dt_kernel(const float* __restrict__ zx,
                               const float* __restrict__ cw,
                               const float* __restrict__ cb,
                               const float* __restrict__ dtbias,
                               float* __restrict__ xBC,
                               float* __restrict__ dtO, int total) {
  int i = blockIdx.x * blockDim.x + threadIdx.x;
  int stride = gridDim.x * blockDim.x;
  for (; i < total; i += stride) {
    int t = i / 1296;
    int c = i - t * 1296;
    if (c < CONV_DIM) {
      int b = t / SEQ, l = t - b * SEQ;
      float acc = cb[c];
#pragma unroll
      for (int k = 0; k < 4; ++k) {
        int ls = l + k - 3;
        if (ls >= 0)
          acc = fmaf(zx[(size_t)(b * SEQ + ls) * D_PROJ_PAD + D_INNER + c],
                     cw[c * 4 + k], acc);
      }
      xBC[(size_t)t * CONV_DIM + c] = silu_f(acc);
    } else {
      int hh = c - CONV_DIM;
      float v = zx[(size_t)t * D_PROJ_PAD + D_INNER + CONV_DIM + hh] + dtbias[hh];
      dtO[t * NHEADS + hh] = (v > 20.f) ? v : log1pf(__expf(v));
    }
  }
}

// ---------------------------------------------------------------------------
// SSD scan. One block per (batch, head). 256 threads: thread = (p, g),
// p = row 0..63, g = n-group 0..3 (32 n-columns each, state in 32 VGPRs).
// Double-buffered LDS staging of x/B/C per timestep, one barrier per step.
// ---------------------------------------------------------------------------
__device__ __forceinline__ void scan_stage(float* sx, float* sBv, float* sCv,
                                           const float* __restrict__ xr,
                                           int h, int tid) {
  if (tid < 64)        sx[tid]        = xr[h * HEADDIM + tid];
  else if (tid < 192)  sBv[tid - 64]  = xr[D_INNER + (tid - 64)];
  else                 sCv[tid - 192] = xr[D_INNER + D_STATE + (tid - 192)];
  if (tid < 64)        sCv[64 + tid]  = xr[D_INNER + D_STATE + 64 + tid];
}

__global__ __launch_bounds__(256) void ssd_scan_kernel(
    const float* __restrict__ xBC, const float* __restrict__ dtv,
    const float* __restrict__ A_log, const float* __restrict__ Dp,
    float* __restrict__ yout) {
  const int bh = blockIdx.x;
  const int b = bh >> 4, h = bh & 15;
  const float Ah = -__expf(A_log[h]);
  const float Dh = Dp[h];

  __shared__ float sx[2][64];
  __shared__ float sBv[2][128];
  __shared__ float sCv[2][128];

  const int tid = threadIdx.x;
  const int p = tid >> 2;
  const int g = tid & 3;
  const int nb = g * 32;

  float s[32];
#pragma unroll
  for (int j = 0; j < 32; ++j) s[j] = 0.f;

  const float* x0 = xBC + (size_t)(b * SEQ) * CONV_DIM;
  scan_stage(sx[0], sBv[0], sCv[0], x0, h, tid);

  for (int l = 0; l < SEQ; ++l) {
    const int cur = l & 1;
    __syncthreads();
    if (l + 1 < SEQ)
      scan_stage(sx[cur ^ 1], sBv[cur ^ 1], sCv[cur ^ 1],
                 x0 + (size_t)(l + 1) * CONV_DIM, h, tid);

    const size_t tix = (size_t)(b * SEQ + l);
    const float dt = dtv[tix * NHEADS + h];
    const float dA = __expf(dt * Ah);
    const float xp = sx[cur][p];
    const float coef = dt * xp;
    float acc = 0.f;
#pragma unroll
    for (int j = 0; j < 32; ++j) {
      float Bn = sBv[cur][nb + j];
      float Cn = sCv[cur][nb + j];
      s[j] = fmaf(coef, Bn, dA * s[j]);
      acc = fmaf(s[j], Cn, acc);
    }
    acc += __shfl_xor(acc, 1, 32);
    acc += __shfl_xor(acc, 2, 32);
    if (g == 0)
      yout[tix * D_INNER + h * HEADDIM + p] = acc + Dh * xp;
  }
}

// ---------------------------------------------------------------------------
// Per-token cross-entropy from logits (vocab = 256 = blockDim)
// ---------------------------------------------------------------------------
__global__ __launch_bounds__(256) void loss_token_kernel(
    const float* __restrict__ logits, const int* __restrict__ targets,
    float* __restrict__ lossb) {
  const int t = blockIdx.x, tid = threadIdx.x;
  float v = logits[(size_t)t * VOCAB + tid];
  float m = v;
#pragma unroll
  for (int off = 16; off >= 1; off >>= 1) m = fmaxf(m, __shfl_xor(m, off, 32));
  __shared__ float sm[8];
  __shared__ float s2[8];
  if ((tid & 31) == 0) sm[tid >> 5] = m;
  __syncthreads();
  float mm = sm[0];
#pragma unroll
  for (int j = 1; j < 8; ++j) mm = fmaxf(mm, sm[j]);
  float e = __expf(v - mm);
#pragma unroll
  for (int off = 16; off >= 1; off >>= 1) e += __shfl_xor(e, off, 32);
  if ((tid & 31) == 0) s2[tid >> 5] = e;
  __syncthreads();
  if (tid == 0) {
    float tot = 0.f;
#pragma unroll
    for (int j = 0; j < 8; ++j) tot += s2[j];
    int tgt = targets[t];
    float lt = logits[(size_t)t * VOCAB + tgt];
    lossb[t] = -(lt - mm - __logf(tot));
  }
}

__global__ __launch_bounds__(256) void loss_reduce_kernel(
    const float* __restrict__ lossb, float* __restrict__ out, int T) {
  const int tid = threadIdx.x;
  float s = 0.f;
  for (int i = tid; i < T; i += 256) s += lossb[i];
#pragma unroll
  for (int off = 16; off >= 1; off >>= 1) s += __shfl_xor(s, off, 32);
  __shared__ float sw[8];
  if ((tid & 31) == 0) sw[tid >> 5] = s;
  __syncthreads();
  if (tid == 0) {
    float tot = 0.f;
#pragma unroll
    for (int j = 0; j < 8; ++j) tot += sw[j];
    out[0] = tot / (float)T;
  }
}

// ---------------------------------------------------------------------------
// Host orchestration
// ---------------------------------------------------------------------------
extern "C" void kernel_launch(void* const* d_in, const int* in_sizes, int n_in,
                              void* d_out, int out_size, void* d_ws, size_t ws_size,
                              hipStream_t stream) {
  (void)in_sizes; (void)n_in; (void)out_size; (void)ws_size;

  const int*   tokens      = (const int*)d_in[0];
  const int*   targets     = (const int*)d_in[1];
  const float* embedding   = (const float*)d_in[2];
  const float* in_proj_w   = (const float*)d_in[3];
  const float* conv_w      = (const float*)d_in[4];
  const float* conv_b      = (const float*)d_in[5];
  const float* dt_bias     = (const float*)d_in[6];
  const float* A_log       = (const float*)d_in[7];
  const float* Dp          = (const float*)d_in[8];
  const float* gnorm_w     = (const float*)d_in[9];
  const float* out_proj_w  = (const float*)d_in[10];
  const float* block_norm_w= (const float*)d_in[11];
  const float* norm_f_w    = (const float*)d_in[12];

  const size_t T = TOKENS;

  // workspace carve (256B aligned)
  size_t off = 0;
  auto carve = [&](size_t bytes) -> char* {
    char* p = (char*)d_ws + off;
    off += (bytes + 255) & ~(size_t)255;
    return p;
  };
  u16*   w_in_bf  = (u16*)  carve((size_t)N_LAYER * D_PROJ_PAD * D_MODEL * 2);
  u16*   w_out_bf = (u16*)  carve((size_t)N_LAYER * D_MODEL * D_INNER * 2);
  u16*   emb_bf   = (u16*)  carve((size_t)VOCAB * D_MODEL * 2);
  float* hidden   = (float*)carve(T * D_MODEL * 4);
  float* resid    = (float*)carve(T * D_MODEL * 4);
  u16*   xbf      = (u16*)  carve(T * D_INNER * 2);   // bf16 GEMM input
  float* zx       = (float*)carve(T * D_PROJ_PAD * 4);
  float* xBC      = (float*)carve(T * CONV_DIM * 4);
  float* dtb      = (float*)carve(T * NHEADS * 4);
  float* ybuf     = (float*)carve(T * D_INNER * 4);
  float* lossb    = (float*)carve(T * 4);

  float* d_logits = (float*)d_out;
  float* d_loss   = d_logits + T * VOCAB;

  // weight conversion (every call; deterministic)
  {
    long long n1 = (long long)N_LAYER * D_PROJ_PAD * D_MODEL;
    long long n2 = (long long)N_LAYER * D_MODEL * D_INNER;
    long long n3 = (long long)VOCAB * D_MODEL;
    cvt_inproj_pad_kernel<<<(unsigned)((n1 + 255) / 256), 256, 0, stream>>>(in_proj_w, w_in_bf, n1);
    cvt_bf16_kernel<<<(unsigned)((n2 + 255) / 256), 256, 0, stream>>>(out_proj_w, w_out_bf, n2);
    cvt_bf16_kernel<<<(unsigned)((n3 + 255) / 256), 256, 0, stream>>>(embedding,  emb_bf,   n3);
  }

  // embedding + zero residual
  {
    int total = (int)(T * D_MODEL);
    embed_kernel<<<(total + 255) / 256, 256, 0, stream>>>(tokens, embedding, hidden, resid, total);
  }

  const dim3 gemm_in_grid(D_PROJ_PAD / 64, (unsigned)(T / 128));   // 37 x 96
  const dim3 gemm_out_grid(D_MODEL / 64,   (unsigned)(T / 128));   // 8 x 96
  const dim3 gemm_lg_grid (VOCAB / 64,     (unsigned)(T / 128));   // 4 x 96
  const int conv_total = (int)(T * (CONV_DIM + NHEADS));

  for (int l = 0; l < N_LAYER; ++l) {
    // res += h ; hn = bf16(rms(res)*block_norm_w)
    addnorm_kernel<<<(unsigned)T, 256, 0, stream>>>(
        hidden, resid, block_norm_w + (size_t)l * D_MODEL, xbf, D_MODEL);

    // zx = hn @ in_proj_w^T        [T,512] x [2368,512]^T (padded)
    gemm_bf16_wmma<<<gemm_in_grid, 256, 0, stream>>>(
        xbf, w_in_bf + (size_t)l * D_PROJ_PAD * D_MODEL, zx, D_PROJ_PAD, D_MODEL);

    // conv + silu + softplus(dt)
    conv_dt_kernel<<<(conv_total + 255) / 256, 256, 0, stream>>>(
        zx, conv_w + (size_t)l * CONV_DIM * 4, conv_b + (size_t)l * CONV_DIM,
        dt_bias + (size_t)l * NHEADS, xBC, dtb, conv_total);

    // SSD recurrence (one block per batch*head)
    ssd_scan_kernel<<<BATCH * NHEADS, 256, 0, stream>>>(
        xBC, dtb, A_log + (size_t)l * NHEADS, Dp + (size_t)l * NHEADS, ybuf);

    // gated RMSNorm -> bf16
    gatenorm_kernel<<<(unsigned)T, 256, 0, stream>>>(
        ybuf, zx, gnorm_w + (size_t)l * D_INNER, xbf);

    // hidden = yg @ out_proj_w^T   [T,1024] x [512,1024]^T
    gemm_bf16_wmma<<<gemm_out_grid, 256, 0, stream>>>(
        xbf, w_out_bf + (size_t)l * D_MODEL * D_INNER, hidden, D_MODEL, D_INNER);
  }

  // final norm, logits, loss
  addnorm_kernel<<<(unsigned)T, 256, 0, stream>>>(hidden, resid, norm_f_w, xbf, D_MODEL);
  gemm_bf16_wmma<<<gemm_lg_grid, 256, 0, stream>>>(xbf, emb_bf, d_logits, VOCAB, D_MODEL);
  loss_token_kernel<<<(unsigned)T, 256, 0, stream>>>(d_logits, targets, lossb);
  loss_reduce_kernel<<<1, 256, 0, stream>>>(lossb, d_loss, (int)T);
}
